// TransformerBlock_7009386627369
// MI455X (gfx1250) — matrix operand order repaired and checked by reading, not compiled, not run on hardware
//
#include <hip/hip_runtime.h>
#include <hip/hip_bf16.h>
#include <math.h>
#include <stdint.h>

typedef __attribute__((ext_vector_type(16))) _Float16 v16h;
typedef __attribute__((ext_vector_type(8)))  _Float16 v8h;
typedef __attribute__((ext_vector_type(8)))  float    v8f;
typedef __attribute__((ext_vector_type(2)))  int      v2i;

#define TB 2048
#define CB 1024
#define HB 16
#define DB 64
#define BB 2

// ---------- CDNA5 async-copy plumbing (guarded; falls back to plain copies) ----------

#if defined(__gfx1250__) && __has_builtin(__builtin_amdgcn_global_load_async_to_lds_b64) && \
    __has_builtin(__builtin_amdgcn_s_wait_asynccnt)
#define USE_ASYNC 1
#else
#define USE_ASYNC 0
#endif

#define AS1 __attribute__((address_space(1)))
#define AS3 __attribute__((address_space(3)))

__device__ __forceinline__ void copy8h_async(const _Float16* g, _Float16* l) {
#if USE_ASYNC
  // builtin proto (from clang diagnostic): (v2i AS1* src, v2i AS3* dst, imm off, cpol)
  // generic->AS casts must go through integers; LDS generic addr low 32 bits = LDS offset.
  __builtin_amdgcn_global_load_async_to_lds_b64(
      (AS1 v2i*)(uintptr_t)g,
      (AS3 v2i*)(unsigned int)(uintptr_t)l,
      0, 0);
#else
  *(uint64_t*)l = *(const uint64_t*)g;
#endif
}
__device__ __forceinline__ void async_join() {
#if USE_ASYNC
  __builtin_amdgcn_s_wait_asynccnt(0);
#endif
}

// ---------- helpers ----------

__device__ __forceinline__ float half_red_max(float v) {
  #pragma unroll
  for (int off = 1; off < 16; off <<= 1) v = fmaxf(v, __shfl_xor(v, off, 32));
  return v;
}
__device__ __forceinline__ float half_red_sum(float v) {
  #pragma unroll
  for (int off = 1; off < 16; off <<= 1) v += __shfl_xor(v, off, 32);
  return v;
}

// LIF rate via hardware rcp/log (I_safe > 1 so -1/I in (-1,0): log argument in (0,1))
__device__ __forceinline__ float lif_rate(float I) {
  bool good = I > 1.0f + 1e-7f;
  float Is = good ? I : 2.0f;
  float r = __builtin_amdgcn_rcpf(0.002f - 0.02f * __logf(1.0f - __builtin_amdgcn_rcpf(Is)));
  return good ? r : 0.0f;
}

// ---------- elementwise f32 -> f16 convert (weights / enc, once per launch) ----------

__global__ void cvt_f32_f16(const float* __restrict__ src, _Float16* __restrict__ dst) {
  size_t i = ((size_t)blockIdx.x * 256 + threadIdx.x) * 8;
  #pragma unroll
  for (int j = 0; j < 8; ++j) dst[i + j] = (_Float16)src[i + j];
}

// ---------- LayerNorm: one block (256 thr) per row of C=1024, f16 output ----------

__global__ void ln_kernel(const float* __restrict__ x, const float* __restrict__ g,
                          const float* __restrict__ bta, _Float16* __restrict__ y) {
  int row = blockIdx.x, tid = threadIdx.x;
  const float* xr = x + (size_t)row * CB;
  _Float16* yr = y + (size_t)row * CB;
  float v[4];
  float s = 0.f;
  #pragma unroll
  for (int i = 0; i < 4; ++i) { v[i] = xr[tid + i * 256]; s += v[i]; }
  __shared__ float red[8];
  #pragma unroll
  for (int off = 1; off < 32; off <<= 1) s += __shfl_xor(s, off, 32);
  if ((tid & 31) == 0) red[tid >> 5] = s;
  __syncthreads();
  float tot = 0.f;
  #pragma unroll
  for (int i = 0; i < 8; ++i) tot += red[i];
  float mean = tot * (1.0f / CB);
  float vs = 0.f;
  #pragma unroll
  for (int i = 0; i < 4; ++i) { float d = v[i] - mean; vs += d * d; }
  #pragma unroll
  for (int off = 1; off < 32; off <<= 1) vs += __shfl_xor(vs, off, 32);
  __syncthreads();
  if ((tid & 31) == 0) red[tid >> 5] = vs;
  __syncthreads();
  float vtot = 0.f;
  #pragma unroll
  for (int i = 0; i < 8; ++i) vtot += red[i];
  float inv = rsqrtf(vtot * (1.0f / CB) + 1e-5f);
  #pragma unroll
  for (int i = 0; i < 4; ++i) {
    int c = tid + i * 256;
    yr[c] = (_Float16)((v[i] - mean) * inv * g[c] + bta[c]);
  }
}

// ---------- WMMA GEMM: out[M,N] = A[M,K] @ W[K,N] + bias (+epilogue) ----------
// A,W are f16. Block 256 thr = 8 waves; tile 64x128; wave tile 16x64.
// A tile staged with async global->LDS b64 copies; W tile transposed manually.
// EPI: 0=none, 1=exact GELU, 2=residual add.  OUTH: 1 = store f16.

template <int EPI, int OUTH>
__global__ void gemm_wmma_kernel(const _Float16* __restrict__ A, const _Float16* __restrict__ W,
                                 const float* __restrict__ bias, const float* __restrict__ res,
                                 float* __restrict__ out, int M, int N, int K) {
  __shared__ _Float16 As[64 * 40];    // 64 rows x 32 k, stride 40 (16B aligned rows)
  __shared__ _Float16 Bs[128 * 40];   // W tile stored transposed: [n][k]
  int tid = threadIdx.x;
  int lane = tid & 31, wid = tid >> 5;
  int mr = wid & 3, ng = wid >> 2;            // 4 row-groups x 2 col-groups
  int ln16 = lane & 15, hi = lane >> 4;       // half-wave id
  int rowBase = blockIdx.y * 64;
  int colBase = blockIdx.x * 128;
  v8f acc[4] = {};

  int ar = tid >> 3, ac8 = (tid & 7) * 4;     // A staging: 2 rows x 8B per thread
  int wk = tid >> 3, wn0 = (tid & 7) * 16;    // W staging assignment

  for (int k0 = 0; k0 < K; k0 += 32) {
    // stage A tile (64x32 f16) via async copies
    const _Float16* ap0 = A + (size_t)(rowBase + ar) * K + k0 + ac8;
    const _Float16* ap1 = ap0 + (size_t)32 * K;
    copy8h_async(ap0, &As[ar * 40 + ac8]);
    copy8h_async(ap1, &As[(ar + 32) * 40 + ac8]);
    // stage W tile transposed (32x128 f16 -> [n][k])
    const _Float16* wp = W + (size_t)(k0 + wk) * N + colBase + wn0;
    #pragma unroll
    for (int j = 0; j < 16; ++j) Bs[(wn0 + j) * 40 + wk] = wp[j];
    if (k0 + 32 < K) {
      __builtin_prefetch(ap0 + 32, 0, 0);
      __builtin_prefetch(wp + (size_t)32 * N, 0, 0);
    }
    async_join();
    __syncthreads();

    // A fragment: lane holds row mr*16+ln16; K halves per ISA 16-bit A layout
    int arow = mr * 16 + ln16;
    v16h afrag;
    {
      v8h lo = *(const v8h*)&As[arow * 40 + hi * 8];
      v8h hi8 = *(const v8h*)&As[arow * 40 + 16 + hi * 8];
      #pragma unroll
      for (int j = 0; j < 8; ++j) { afrag[j] = lo[j]; afrag[j + 8] = hi8[j]; }
    }
    #pragma unroll
    for (int s = 0; s < 4; ++s) {
      int bn = ng * 64 + s * 16 + ln16;       // B lane: column n, K halves per ISA 16-bit layout
      v16h bfrag;
      v8h l0 = *(const v8h*)&Bs[bn * 40 + hi * 8];
      v8h l1 = *(const v8h*)&Bs[bn * 40 + 16 + hi * 8];
      #pragma unroll
      for (int j = 0; j < 8; ++j) { bfrag[j] = l0[j]; bfrag[j + 8] = l1[j]; }
      acc[s] = __builtin_amdgcn_wmma_f32_16x16x32_f16(false, afrag, false, bfrag,
                                                      (short)0, acc[s], false, false);
    }
    __syncthreads();
  }

  // epilogue
  #pragma unroll
  for (int s = 0; s < 4; ++s) {
    int col = colBase + ng * 64 + s * 16 + ln16;
    float bcol = bias[col];
    #pragma unroll
    for (int r = 0; r < 8; ++r) {
      int row = rowBase + mr * 16 + r + hi * 8;
      float v = acc[s][r] + bcol;
      if (EPI == 1) v = 0.5f * v * (1.0f + erff(v * 0.70710678118f));
      if (EPI == 2) v += res[(size_t)row * N + col];
      if (OUTH) ((_Float16*)out)[(size_t)row * N + col] = (_Float16)v;
      else      out[(size_t)row * N + col] = v;
    }
  }
}

// ---------- Attention: flash-style, block-cooperative ----------
// One block = one (b,h) and 128 consecutive queries (8 waves x 16-query tiles).
// E chunk (f16, pre-converted) staged via async copies; V chunk transposed
// manually (f32->f16).  scores = LIF(G*(Q.E^T)+bias)/8, causal, online softmax, P@V.

__global__ void attn_kernel(const float* __restrict__ qkv, const _Float16* __restrict__ enc16,
                            const float* __restrict__ gain, const float* __restrict__ battn,
                            _Float16* __restrict__ outp) {
  __shared__ _Float16 Es[32 * 72];            // E chunk  [key][d],  stride 72
  __shared__ _Float16 Vt[64 * 40];            // V chunk  [d][key],  stride 40
  __shared__ _Float16 Pl[8][16 * 40];         // per-wave P transpose tile
  int tid = threadIdx.x, lane = tid & 31, wid = tid >> 5;
  int ln16 = lane & 15, hi = lane >> 4;

  const int G = TB / 128;                     // query groups per (b,h)
  int blk = blockIdx.x;                       // B*H*G blocks
  int b = blk / (HB * G);
  int rem = blk % (HB * G);
  int h = rem / G;
  int t0g = (rem % G) * 128;                  // block query base
  int t0 = t0g + wid * 16;                    // wave query tile base

  const size_t row3C = (size_t)3 * CB;
  const float* Qb = qkv + (size_t)b * TB * row3C + (size_t)h * DB;
  const float* Vb = qkv + (size_t)b * TB * row3C + (size_t)2 * CB + (size_t)h * DB;
  const _Float16* Eh = enc16 + (size_t)h * TB * DB;

  // Q fragments for two K=32 steps (A layout: lane = query row t0+ln16)
  v16h qf[2];
  #pragma unroll
  for (int ks = 0; ks < 2; ++ks) {
    const float* qp = Qb + (size_t)(t0 + ln16) * row3C + ks * 32 + hi * 8;
    #pragma unroll
    for (int j = 0; j < 8; ++j) { qf[ks][j] = (_Float16)qp[j]; qf[ks][j + 8] = (_Float16)qp[16 + j]; }
  }
  // per-row gain/bias (row = t0 + r + hi*8 per C-layout)
  float gv[8], bv[8], m[8], l[8];
  #pragma unroll
  for (int r = 0; r < 8; ++r) {
    int t = t0 + r + hi * 8;
    gv[r] = gain[(size_t)h * TB + t];
    bv[r] = battn[(size_t)h * TB + t];
    m[r] = -1e30f; l[r] = 0.f;
  }
  v8f acc[4] = {};
  _Float16* P = &Pl[wid][0];

  int sr = tid >> 3, sc0 = (tid & 7) * 8;     // V staging (32 rows x 8 f32)
  int ec8 = (tid & 7) * 4;                    // E staging (2x 8B per thread)

  int nChunks = (t0g + 127) / 32 + 1;         // uniform across the block
  for (int kc = 0; kc < nChunks; ++kc) {
    int s0 = kc * 32;
    // ---- cooperative staging ----
    {
      const _Float16* ep = Eh + (size_t)(s0 + sr) * DB + ec8;
      copy8h_async(ep, &Es[sr * 72 + ec8]);
      copy8h_async(ep + 32, &Es[sr * 72 + 32 + ec8]);
      const float* vp = Vb + (size_t)(s0 + sr) * row3C + sc0;
      #pragma unroll
      for (int j = 0; j < 8; ++j) Vt[(sc0 + j) * 40 + sr] = (_Float16)vp[j];
      if (kc + 1 < nChunks) {
        __builtin_prefetch(ep + 32 * DB, 0, 0);
        __builtin_prefetch(vp + 32 * row3C, 0, 0);
      }
    }
    async_join();
    __syncthreads();

    if (s0 <= t0 + 15) {                      // wave-uniform: EXEC stays all-ones
      // ---- score tiles: 2 key-subtiles x 2 K-steps ----
      v8f sc[2] = {};
      #pragma unroll
      for (int n0 = 0; n0 < 2; ++n0) {
        #pragma unroll
        for (int ks = 0; ks < 2; ++ks) {
          const _Float16* eb = &Es[(n0 * 16 + ln16) * 72 + ks * 32 + hi * 8];
          v8h e0 = *(const v8h*)eb;
          v8h e1 = *(const v8h*)(eb + 16);
          v16h ef;
          #pragma unroll
          for (int j = 0; j < 8; ++j) { ef[j] = e0[j]; ef[j + 8] = e1[j]; }
          sc[n0] = __builtin_amdgcn_wmma_f32_16x16x32_f16(false, qf[ks], false, ef,
                                                          (short)0, sc[n0], false, false);
        }
      }
      // ---- gain/bias + LIF + causal mask (in place) ----
      #pragma unroll
      for (int n0 = 0; n0 < 2; ++n0) {
        int key = s0 + n0 * 16 + ln16;
        #pragma unroll
        for (int r = 0; r < 8; ++r) {
          int t = t0 + r + hi * 8;
          float I = gv[r] * sc[n0][r] + bv[r];
          float rate = lif_rate(I) * 0.125f;  // 1/sqrt(D)
          sc[n0][r] = (key <= t) ? rate : -1e30f;
        }
      }
      // ---- online softmax update ----
      #pragma unroll
      for (int r = 0; r < 8; ++r) {
        float cmax = half_red_max(fmaxf(sc[0][r], sc[1][r]));
        float mn = fmaxf(m[r], cmax);
        float scale = __expf(m[r] - mn);
        m[r] = mn;
        float p0 = __expf(sc[0][r] - mn);
        float p1 = __expf(sc[1][r] - mn);
        P[(r + hi * 8) * 40 + ln16] = (_Float16)p0;
        P[(r + hi * 8) * 40 + 16 + ln16] = (_Float16)p1;
        float psum = half_red_sum(p0 + p1);
        l[r] = l[r] * scale + psum;
        #pragma unroll
        for (int s = 0; s < 4; ++s) acc[s][r] *= scale;
      }
      // ---- P as A-fragment (same-wave LDS transpose) ----
      v16h pf;
      {
        const _Float16* pr = &P[ln16 * 40 + hi * 8];
        #pragma unroll
        for (int j = 0; j < 8; ++j) { pf[j] = pr[j]; pf[j + 8] = pr[16 + j]; }
      }
      // ---- P @ V (B-frags from transposed LDS tile) ----
      #pragma unroll
      for (int s = 0; s < 4; ++s) {
        const _Float16* vb2 = &Vt[(s * 16 + ln16) * 40 + hi * 8];
        v8h w0 = *(const v8h*)vb2;
        v8h w1 = *(const v8h*)(vb2 + 16);
        v16h vf;
        #pragma unroll
        for (int j = 0; j < 8; ++j) { vf[j] = w0[j]; vf[j + 8] = w1[j]; }
        acc[s] = __builtin_amdgcn_wmma_f32_16x16x32_f16(false, pf, false, vf,
                                                        (short)0, acc[s], false, false);
      }
    }
    __syncthreads();
  }
  // ---- write O * (1/l) as f16 ----
  #pragma unroll
  for (int r = 0; r < 8; ++r) {
    int t = t0 + r + hi * 8;
    float linv = __builtin_amdgcn_rcpf(l[r]);
    #pragma unroll
    for (int s = 0; s < 4; ++s) {
      outp[((size_t)b * TB + t) * CB + h * DB + s * 16 + ln16] = (_Float16)(acc[s][r] * linv);
    }
  }
}

// ---------- launcher ----------

#define MB(x) ((size_t)(x) * 1024 * 1024)

extern "C" void kernel_launch(void* const* d_in, const int* in_sizes, int n_in,
                              void* d_out, int out_size, void* d_ws, size_t ws_size,
                              hipStream_t stream) {
  const float* x      = (const float*)d_in[0];
  const float* ln1_g  = (const float*)d_in[1];
  const float* ln1_b  = (const float*)d_in[2];
  const float* qkv_w  = (const float*)d_in[3];
  const float* qkv_b  = (const float*)d_in[4];
  const float* out_w  = (const float*)d_in[5];
  const float* out_b  = (const float*)d_in[6];
  const float* ln2_g  = (const float*)d_in[7];
  const float* ln2_b  = (const float*)d_in[8];
  const float* mlp_w1 = (const float*)d_in[9];
  const float* mlp_b1 = (const float*)d_in[10];
  const float* mlp_w2 = (const float*)d_in[11];
  const float* mlp_b2 = (const float*)d_in[12];
  const float* enc    = (const float*)d_in[13];
  const float* gaini  = (const float*)d_in[14];
  const float* biasa  = (const float*)d_in[15];
  float* out = (float*)d_out;

  const int M = BB * TB;                                   // 4096
  char* ws = (char*)d_ws;
  _Float16* xh   = (_Float16*)(ws);                        // 8 MB  LN1 out (f16)
  float*    qkvb = (float*)   (ws + MB(8));                // 48 MB QKV (f32)
  _Float16* ah   = (_Float16*)(ws + MB(56));               // 8 MB  attn out (f16)
  float*    x2   = (float*)   (ws + MB(64));               // 16 MB residual 1 (f32)
  _Float16* hh   = (_Float16*)(ws + MB(80));               // 8 MB  LN2 out (f16)
  _Float16* mhh  = (_Float16*)(ws + MB(88));               // 32 MB MLP hidden (f16)
  _Float16* wq16 = (_Float16*)(ws + MB(120));              // 6 MB
  _Float16* wo16 = (_Float16*)(ws + MB(126));              // 2 MB
  _Float16* w116 = (_Float16*)(ws + MB(128));              // 8 MB
  _Float16* w216 = (_Float16*)(ws + MB(136));              // 8 MB
  _Float16* e16  = (_Float16*)(ws + MB(144));              // 4 MB

  // 0. one-time f32->f16 conversions (weights + encoder)
  cvt_f32_f16<<<(CB * 3 * CB) / 2048, 256, 0, stream>>>(qkv_w, wq16);
  cvt_f32_f16<<<(CB * CB) / 2048, 256, 0, stream>>>(out_w, wo16);
  cvt_f32_f16<<<(CB * 4 * CB) / 2048, 256, 0, stream>>>(mlp_w1, w116);
  cvt_f32_f16<<<(4 * CB * CB) / 2048, 256, 0, stream>>>(mlp_w2, w216);
  cvt_f32_f16<<<(HB * TB * DB) / 2048, 256, 0, stream>>>(enc, e16);

  // 1. LN1 -> f16
  ln_kernel<<<M, 256, 0, stream>>>(x, ln1_g, ln1_b, xh);
  // 2. QKV projection: (M,1024) x (1024,3072) -> f32
  gemm_wmma_kernel<0, 0><<<dim3(3 * CB / 128, M / 64), 256, 0, stream>>>(
      xh, wq16, qkv_b, nullptr, qkvb, M, 3 * CB, CB);
  // 3. LIF attention -> f16
  attn_kernel<<<BB * HB * (TB / 128), 256, 0, stream>>>(qkvb, e16, gaini, biasa, ah);
  // 4. out projection + residual: x2 = x + attn @ out_w + out_b
  gemm_wmma_kernel<2, 0><<<dim3(CB / 128, M / 64), 256, 0, stream>>>(
      ah, wo16, out_b, x, x2, M, CB, CB);
  // 5. LN2 -> f16
  ln_kernel<<<M, 256, 0, stream>>>(x2, ln2_g, ln2_b, hh);
  // 6. MLP up + exact GELU -> f16 hidden
  gemm_wmma_kernel<1, 1><<<dim3(4 * CB / 128, M / 64), 256, 0, stream>>>(
      hh, w116, mlp_b1, nullptr, (float*)mhh, M, 4 * CB, CB);
  // 7. MLP down + residual -> out (f32)
  gemm_wmma_kernel<2, 0><<<dim3(CB / 128, M / 64), 256, 0, stream>>>(
      mhh, w216, mlp_b2, x2, out, M, CB, 4 * CB);
}
